// lstm_model_57578331570197
// MI455X (gfx1250) — matrix operand order repaired and checked by reading, not compiled, not run on hardware
//
#include <hip/hip_runtime.h>

typedef __attribute__((ext_vector_type(16))) _Float16 v16h;
typedef __attribute__((ext_vector_type(8)))  float    v8f;

#define HD   31    // hidden size
#define G4   124   // 4*H gates
#define NT   8     // ceil(124/16) N-tiles

__device__ __forceinline__ float sigm(float z) {
    return 1.0f / (1.0f + __expf(-z));
}

// gate n lives in LDS gate-buffer slot (n/16)*32 + (n%16)
__device__ __forceinline__ int gidx(int n) {
    return ((n >> 4) << 5) + (n & 15);
}

// Build one 32x16 (KxN) f16 B-tile of W^T, where W is G4 x ncols row-major.
// Layout chosen here (contiguous k, same as pack_avec): lanes 0-15 hold K=0..15 (N = lane),
// lanes 16-31 hold K=16..31 (N = lane-16); component c <-> K = kbase + c.
__device__ __forceinline__ v16h build_btile(const float* __restrict__ W,
                                            int ncols, int nb, int lane) {
    int n = nb * 16 + (lane & 15);
    int kbase = (lane < 16) ? 0 : 16;
    v16h b = {};
#pragma unroll
    for (int c = 0; c < 16; ++c) {
        int k = kbase + c;
        float v = 0.0f;
        if (n < G4 && k < ncols) v = W[n * ncols + k];
        b[c] = (_Float16)v;
    }
    return b;
}

// Pack a length-32 (zero-padded) vector from LDS into row 0 of a 16x32 f16
// A-matrix. Row 0 lives in lane 0 (K=0..15 -> comps 0..15) and lane 16
// (K=16..31 -> comps 0..15). All other lanes/rows are zero.
__device__ __forceinline__ v16h pack_avec(const float* hl, int lane) {
    v16h a = {};
    if (lane == 0 || lane == 16) {
        int kb = (lane == 16) ? 16 : 0;
#pragma unroll
        for (int i = 0; i < 8; ++i) a[i]     = (_Float16)hl[kb + i];
#pragma unroll
        for (int i = 0; i < 8; ++i) a[8 + i] = (_Float16)hl[kb + 8 + i];
    }
    return a;
}

__global__ void lstm_wmma_persistent(
    const float* __restrict__ input,
    const float* __restrict__ w_ih_l0, const float* __restrict__ w_hh_l0,
    const float* __restrict__ b_ih_l0, const float* __restrict__ b_hh_l0,
    const float* __restrict__ w_ih_l1, const float* __restrict__ w_hh_l1,
    const float* __restrict__ b_ih_l1, const float* __restrict__ b_hh_l1,
    const float* __restrict__ w_lin,   const float* __restrict__ b_lin,
    const int*   __restrict__ futp,    int T,
    float* __restrict__ out)
{
    const int lane = threadIdx.x;

    // All 24 weight B-tiles live in LDS (24 KB): single wave owns the WGP's
    // full LDS bandwidth; ds_load addresses are loop-invariant so they issue
    // early and overlap the previous activation stage.
    __shared__ v16h  b0t[NT * 32];   // layer-0 W_hh tiles
    __shared__ v16h  b1i[NT * 32];   // layer-1 W_ih tiles
    __shared__ v16h  b1h[NT * 32];   // layer-1 W_hh tiles
    __shared__ float h1l[32];
    __shared__ float h2l[32];
    __shared__ float gl[NT * 32];    // raw WMMA row-0 outputs, slot = tile*32+lane

    // ---- one-time staging ----------------------------------------------
    for (int t8 = 0; t8 < NT; ++t8) {
        b0t[t8 * 32 + lane] = build_btile(w_hh_l0, HD, t8, lane);
        b1i[t8 * 32 + lane] = build_btile(w_ih_l1, HD, t8, lane);
        b1h[t8 * 32 + lane] = build_btile(w_hh_l1, HD, t8, lane);
    }

    // Per-lane constants in the ACTIVATION layout: lane j (< 31) owns hidden
    // unit j and consumes gates n = g*31+j for g = i,f,g,o.  16 floats total,
    // no indexed arrays on the WMMA path -> nothing to spill.
    float bi0[4], bi1[4], wa[4], wb[4];
#pragma unroll
    for (int g = 0; g < 4; ++g) {
        int n = g * HD + lane;
        bool ok = (lane < HD);
        bi0[g] = ok ? (b_ih_l0[n] + b_hh_l0[n]) : 0.0f;
        bi1[g] = ok ? (b_ih_l1[n] + b_hh_l1[n]) : 0.0f;
        wa[g]  = ok ? w_ih_l0[n * 2 + 0] : 0.0f;
        wb[g]  = ok ? w_ih_l0[n * 2 + 1] : 0.0f;
    }
    float wl0 = (lane < HD) ? w_lin[lane]      : 0.0f;
    float wl1 = (lane < HD) ? w_lin[HD + lane] : 0.0f;
    float bl0 = b_lin[0], bl1 = b_lin[1];

    h1l[lane] = 0.0f;
    h2l[lane] = 0.0f;
    __syncthreads();

    v16h a1 = {};            // packed h1 (zero initial state)
    v16h a2 = {};            // packed h2
    const v8f zc = {};       // inline-zero C operand
    float c1 = 0.0f, c2 = 0.0f;
    float y0 = 0.0f, y1 = 0.0f;
    const int total = T + *futp;

    for (int t = 0; t < total; ++t) {
        // x: encoder input, decoder seed (input[T-1]), or fed-back y
        float x0, x1;
        if (t <= T) {
            int ti = (t < T) ? t : (T - 1);
            x0 = input[ti * 2];
            x1 = input[ti * 2 + 1];
        } else {
            x0 = y0;
            x1 = y1;
        }

        // ---- layer 0: raw = W_hh0 @ h1 (bias + W_ih0@x added at activation)
#pragma unroll
        for (int t8 = 0; t8 < NT; ++t8) {
            v8f d = __builtin_amdgcn_wmma_f32_16x16x32_f16(
                false, a1, false, b0t[t8 * 32 + lane], (short)0, zc,
                false, false);
            gl[t8 * 32 + lane] = d[0];   // lanes 16-31 write unread slots
        }
        __syncthreads();
        {
            float h1v = 0.0f;
            if (lane < HD) {
                float gi = gl[gidx(0 * HD + lane)] + bi0[0] + wa[0] * x0 + wb[0] * x1;
                float gf = gl[gidx(1 * HD + lane)] + bi0[1] + wa[1] * x0 + wb[1] * x1;
                float gg = gl[gidx(2 * HD + lane)] + bi0[2] + wa[2] * x0 + wb[2] * x1;
                float go = gl[gidx(3 * HD + lane)] + bi0[3] + wa[3] * x0 + wb[3] * x1;
                c1  = sigm(gf) * c1 + sigm(gi) * tanhf(gg);
                h1v = sigm(go) * tanhf(c1);
            }
            h1l[lane] = h1v;             // lanes >= 31 keep the zero pad
        }
        __syncthreads();
        a1 = pack_avec(h1l, lane);

        // ---- layer 1: raw = W_ih1 @ h1_new + W_hh1 @ h2 ----------------
#pragma unroll
        for (int t8 = 0; t8 < NT; ++t8) {
            v8f d = __builtin_amdgcn_wmma_f32_16x16x32_f16(
                false, a1, false, b1i[t8 * 32 + lane], (short)0, zc,
                false, false);
            d = __builtin_amdgcn_wmma_f32_16x16x32_f16(
                false, a2, false, b1h[t8 * 32 + lane], (short)0, d,
                false, false);
            gl[t8 * 32 + lane] = d[0];
        }
        __syncthreads();
        {
            float h2v = 0.0f;
            if (lane < HD) {
                float gi = gl[gidx(0 * HD + lane)] + bi1[0];
                float gf = gl[gidx(1 * HD + lane)] + bi1[1];
                float gg = gl[gidx(2 * HD + lane)] + bi1[2];
                float go = gl[gidx(3 * HD + lane)] + bi1[3];
                c2  = sigm(gf) * c2 + sigm(gi) * tanhf(gg);
                h2v = sigm(go) * tanhf(c2);
            }
            h2l[lane] = h2v;

            // linear head: y = w_lin @ h2 + b_lin (wave32 tree reduction)
            float p0 = wl0 * h2v;
            float p1 = wl1 * h2v;
#pragma unroll
            for (int off = 16; off > 0; off >>= 1) {
                p0 += __shfl_down(p0, off);
                p1 += __shfl_down(p1, off);
            }
            float yy0 = p0 + bl0;
            float yy1 = p1 + bl1;
            y0 = __shfl(yy0, 0);
            y1 = __shfl(yy1, 0);
            if (lane == 0) {
                out[2 * t]     = y0;
                out[2 * t + 1] = y1;
            }
        }
        __syncthreads();
        a2 = pack_avec(h2l, lane);
    }
}

extern "C" void kernel_launch(void* const* d_in, const int* in_sizes, int n_in,
                              void* d_out, int out_size, void* d_ws, size_t ws_size,
                              hipStream_t stream) {
    (void)n_in; (void)d_ws; (void)ws_size; (void)out_size;
    const float* input   = (const float*)d_in[0];
    const float* w_ih_l0 = (const float*)d_in[1];
    const float* w_hh_l0 = (const float*)d_in[2];
    const float* b_ih_l0 = (const float*)d_in[3];
    const float* b_hh_l0 = (const float*)d_in[4];
    const float* w_ih_l1 = (const float*)d_in[5];
    const float* w_hh_l1 = (const float*)d_in[6];
    const float* b_ih_l1 = (const float*)d_in[7];
    const float* b_hh_l1 = (const float*)d_in[8];
    const float* w_lin   = (const float*)d_in[9];
    const float* b_lin   = (const float*)d_in[10];
    const int*   future  = (const int*)d_in[11];
    int T = in_sizes[0] / 2;   // input is (T, 2) float32

    // Single persistent wave32: the recurrence is strictly sequential, all
    // weights fit on one WGP, WMMA requires whole-wave execution.
    lstm_wmma_persistent<<<1, 32, 0, stream>>>(
        input, w_ih_l0, w_hh_l0, b_ih_l0, b_hh_l0,
        w_ih_l1, w_hh_l1, b_ih_l1, b_hh_l1,
        w_lin, b_lin, future, T, (float*)d_out);
}
